// Alignment_60945585930640
// MI455X (gfx1250) — matrix operand order repaired and checked by reading, not compiled, not run on hardware
//
#include <hip/hip_runtime.h>
#include <hip/hip_bf16.h>
#include <math.h>

#define B_   16
#define LA_  1024
#define LB_  1024
#define H_   512
#define NEGV (-10000000.0f)

typedef __attribute__((ext_vector_type(16))) _Float16 v16h;
typedef __attribute__((ext_vector_type(4)))  _Float16 v4h;
typedef __attribute__((ext_vector_type(8)))  float    v8f;
typedef __attribute__((ext_vector_type(4)))  int      v4i;

// ---------------- async global->LDS (gfx1250) with safe fallback ------------
#if defined(__has_builtin)
#  if __has_builtin(__builtin_amdgcn_global_load_async_to_lds_b128)
#    define HAVE_ASYNC 1
#  endif
#endif
#ifndef HAVE_ASYNC
#  define HAVE_ASYNC 0
#endif

#if HAVE_ASYNC
#  if __has_builtin(__builtin_amdgcn_s_wait_asynccnt)
#    define WAIT_ASYNC(n) __builtin_amdgcn_s_wait_asynccnt(n)
#  else
#    define WAIT_ASYNC(n) asm volatile("s_wait_asynccnt %0" :: "i"(n) : "memory")
#  endif
#else
#  define WAIT_ASYNC(n) ((void)0)
#endif

typedef __attribute__((address_space(1))) v4i gbl_v4i;
typedef __attribute__((address_space(3))) v4i lds_v4i;

__device__ __forceinline__ void cp16(_Float16* lds, const _Float16* g) {
#if HAVE_ASYNC
  __builtin_amdgcn_global_load_async_to_lds_b128(
      (gbl_v4i*)(g), (lds_v4i*)(lds), 0, 0);
#else
  *reinterpret_cast<float4*>(lds) = *reinterpret_cast<const float4*>(g);
#endif
}

// ---------------- WMMA fragment helpers (wave32, 16x16x32 f16) --------------
__device__ __forceinline__ v16h frag_a(const _Float16* lds, int lane, int m0) {
  // A is 16x32 f16: lanes 0-15 -> K {0..7,16..23}; lanes 16-31 -> K {8..15,24..31}
  const _Float16* p = lds + (m0 + (lane & 15)) * 32 + ((lane >> 4) << 3);
  v16h f;
#pragma unroll
  for (int h = 0; h < 8; ++h) { f[h] = p[h]; f[h + 8] = p[h + 16]; }
  return f;
}
__device__ __forceinline__ v16h frag_b(const _Float16* lds, int lane, int n0) {
  // B is 32x16 f16: lane%16 = N column; lanes 0-15 -> K {0..7,16..23}; lanes 16-31 -> K {8..15,24..31}
  const _Float16* p = lds + ((lane >> 4) << 3) * 128 + n0 + (lane & 15);
  v16h f;
#pragma unroll
  for (int h = 0; h < 8; ++h) { f[h] = p[h * 128]; f[h + 8] = p[(h + 16) * 128]; }
  return f;
}
__device__ __forceinline__ v8f wmma16(v16h a, v16h b, v8f c) {
  return __builtin_amdgcn_wmma_f32_16x16x32_f16(false, a, false, b, (short)0, c,
                                                false, false);
}

// ---------------- generic f16 GEMM: Out[M,N] = A[M,K] * Bm[K,N] -------------
// MODE 0: Out = mask ? acc*temp : NEG   (attention scores)
// MODE 1: Out = acc                     (feature outputs)
template <int MODE>
__global__ __launch_bounds__(256) void gemm_wmma(
    const _Float16* __restrict__ A, const _Float16* __restrict__ Bm,
    float* __restrict__ Out, int M, int N, int K,
    const int* __restrict__ maskA, const int* __restrict__ maskB,
    const float* __restrict__ temp) {
  __shared__ _Float16 ldsA[2][128 * 32];
  __shared__ _Float16 ldsB[2][32 * 128];

  const int b = blockIdx.z;
  A   += (size_t)b * M * K;
  Bm  += (size_t)b * K * N;
  Out += (size_t)b * M * N;

  const int t    = threadIdx.x;
  const int lane = t & 31;
  const int wave = t >> 5;
  const int wm   = (wave >> 1) * 32;  // 4 waves along M
  const int wn   = (wave & 1) * 64;   // 2 waves along N
  const int rowBase = blockIdx.y * 128;
  const int colBase = blockIdx.x * 128;
  const int NK = K >> 5;

  v8f acc[2][4];
#pragma unroll
  for (int mi = 0; mi < 2; ++mi)
#pragma unroll
    for (int nf = 0; nf < 4; ++nf)
#pragma unroll
      for (int e = 0; e < 8; ++e) acc[mi][nf][e] = 0.0f;

  auto issue = [&](int buf, int ks) {
    // A tile: 128x32 halves = 512 16B chunks; B tile: 32x128 halves likewise.
    int c0 = t, c1 = t + 256;
    int r = c0 >> 2, ch = (c0 & 3) * 8;
    cp16(&ldsA[buf][r * 32 + ch], A + (size_t)(rowBase + r) * K + ks * 32 + ch);
    r = c1 >> 2; ch = (c1 & 3) * 8;
    cp16(&ldsA[buf][r * 32 + ch], A + (size_t)(rowBase + r) * K + ks * 32 + ch);
    r = c0 >> 4; ch = (c0 & 15) * 8;
    cp16(&ldsB[buf][r * 128 + ch], Bm + (size_t)(ks * 32 + r) * N + colBase + ch);
    r = c1 >> 4; ch = (c1 & 15) * 8;
    cp16(&ldsB[buf][r * 128 + ch], Bm + (size_t)(ks * 32 + r) * N + colBase + ch);
  };

  issue(0, 0);
  for (int ks = 0; ks < NK; ++ks) {
    const int buf = ks & 1;
    if (ks + 1 < NK) {      // prefetch next stage into the other buffer
      issue(buf ^ 1, ks + 1);
      WAIT_ASYNC(4);        // wait until only the 4 newest copies are in flight
    } else {
      WAIT_ASYNC(0);
    }
    __syncthreads();

    // Load ALL fragments first (distinct live registers), then issue the
    // 8 WMMAs back-to-back behind a single DS wait.
    v16h af0 = frag_a(ldsA[buf], lane, wm);
    v16h af1 = frag_a(ldsA[buf], lane, wm + 16);
    v16h bf0 = frag_b(ldsB[buf], lane, wn);
    v16h bf1 = frag_b(ldsB[buf], lane, wn + 16);
    v16h bf2 = frag_b(ldsB[buf], lane, wn + 32);
    v16h bf3 = frag_b(ldsB[buf], lane, wn + 48);

    acc[0][0] = wmma16(af0, bf0, acc[0][0]);
    acc[1][0] = wmma16(af1, bf0, acc[1][0]);
    acc[0][1] = wmma16(af0, bf1, acc[0][1]);
    acc[1][1] = wmma16(af1, bf1, acc[1][1]);
    acc[0][2] = wmma16(af0, bf2, acc[0][2]);
    acc[1][2] = wmma16(af1, bf2, acc[1][2]);
    acc[0][3] = wmma16(af0, bf3, acc[0][3]);
    acc[1][3] = wmma16(af1, bf3, acc[1][3]);

    __syncthreads();
  }

  const float tval = (MODE == 0) ? temp[0] : 0.0f;
#pragma unroll
  for (int mi = 0; mi < 2; ++mi) {
#pragma unroll
    for (int nf = 0; nf < 4; ++nf) {
      const int gj     = colBase + wn + nf * 16 + (lane & 15);
      const int giBase = rowBase + wm + mi * 16 + ((lane >> 4) << 3);
      int mb = 1;
      if (MODE == 0) mb = maskB[(size_t)b * N + gj];
#pragma unroll
      for (int d = 0; d < 8; ++d) {
        const int gi = giBase + d;
        float v = acc[mi][nf][d];
        if (MODE == 0) {
          const int ma = maskA[(size_t)b * M + gi];
          v = (ma && mb) ? v * tval : NEGV;
        }
        Out[(size_t)gi * N + gj] = v;
      }
    }
  }
}

// ---------------- helpers: convert / transpose / softmax stats --------------
__global__ __launch_bounds__(256) void f32_to_f16(const float* __restrict__ x,
                                                  _Float16* __restrict__ y,
                                                  int n4) {
  int i = blockIdx.x * 256 + threadIdx.x;
  if (i < n4) {
    float4 v = reinterpret_cast<const float4*>(x)[i];
    v4h h = { (_Float16)v.x, (_Float16)v.y, (_Float16)v.z, (_Float16)v.w };
    reinterpret_cast<v4h*>(y)[i] = h;
  }
}

// x: [batch][R][C] f32  ->  xt: [batch][C][R] f16
__global__ __launch_bounds__(256) void transpose_f16(const float* __restrict__ x,
                                                     _Float16* __restrict__ xt,
                                                     int R, int C) {
  __shared__ _Float16 tile[32][33];
  const int b = blockIdx.z;
  x  += (size_t)b * R * C;
  xt += (size_t)b * R * C;
  const int c0 = blockIdx.x * 32, r0 = blockIdx.y * 32;
  const int tx = threadIdx.x, ty = threadIdx.y;
#pragma unroll
  for (int rr = 0; rr < 4; ++rr)
    tile[ty + rr * 8][tx] = (_Float16)x[(size_t)(r0 + ty + rr * 8) * C + c0 + tx];
  __syncthreads();
#pragma unroll
  for (int rr = 0; rr < 4; ++rr)
    xt[(size_t)(c0 + ty + rr * 8) * R + r0 + tx] = tile[tx][ty + rr * 8];
}

__device__ __forceinline__ void online_upd(float& m, float& s, float x) {
  float M2 = fmaxf(m, x);
  s = s * __expf(m - M2) + __expf(x - M2);
  m = M2;
}

__global__ __launch_bounds__(256) void row_stats(const float* __restrict__ S,
                                                 float* __restrict__ rmax,
                                                 float* __restrict__ rsum) {
  const int row = blockIdx.x, b = blockIdx.y, t = threadIdx.x;
  const float* p = S + ((size_t)b * LA_ + row) * LB_;
  float m = p[t], s = 1.0f;
  for (int j = t + 256; j < LB_; j += 256) online_upd(m, s, p[j]);
  __shared__ float sm[256], ss[256];
  sm[t] = m; ss[t] = s;
  __syncthreads();
  for (int o = 128; o > 0; o >>= 1) {
    if (t < o) {
      float m1 = sm[t], m2 = sm[t + o], s1 = ss[t], s2 = ss[t + o];
      float M2 = fmaxf(m1, m2);
      sm[t] = M2;
      ss[t] = s1 * __expf(m1 - M2) + s2 * __expf(m2 - M2);
    }
    __syncthreads();
  }
  if (t == 0) {
    rmax[(size_t)b * LA_ + row] = sm[0];
    rsum[(size_t)b * LA_ + row] = ss[0];
  }
}

__global__ __launch_bounds__(256) void col_stats(const float* __restrict__ S,
                                                 float* __restrict__ cmax,
                                                 float* __restrict__ csum) {
  const int j = blockIdx.x * 256 + threadIdx.x, b = blockIdx.y;
  const float* p = S + (size_t)b * LA_ * LB_ + j;
  float m = p[0], s = 1.0f;
  for (int i = 1; i < LA_; ++i) online_upd(m, s, p[(size_t)i * LB_]);
  cmax[(size_t)b * LB_ + j] = m;
  csum[(size_t)b * LB_ + j] = s;
}

// Pb[i][j] = row-softmax (axis 2); PaT[j][i] = col-softmax (axis 1) transposed
__global__ __launch_bounds__(256) void probs(
    const float* __restrict__ S, const float* __restrict__ rmax,
    const float* __restrict__ rsum, const float* __restrict__ cmax,
    const float* __restrict__ csum, _Float16* __restrict__ Pb,
    _Float16* __restrict__ PaT) {
  __shared__ _Float16 tile[32][33];
  const int b = blockIdx.z;
  const int j0 = blockIdx.x * 32, i0 = blockIdx.y * 32;
  const int tx = threadIdx.x, ty = threadIdx.y;
  const float* Sb = S + (size_t)b * LA_ * LB_;
#pragma unroll
  for (int rr = 0; rr < 4; ++rr) {
    const int i = i0 + ty + rr * 8, j = j0 + tx;
    const float x = Sb[(size_t)i * LB_ + j];
    const float pb = __expf(x - rmax[(size_t)b * LA_ + i]) / rsum[(size_t)b * LA_ + i];
    Pb[((size_t)b * LA_ + i) * LB_ + j] = (_Float16)pb;
    const float pa = __expf(x - cmax[(size_t)b * LB_ + j]) / csum[(size_t)b * LB_ + j];
    tile[ty + rr * 8][tx] = (_Float16)pa;
  }
  __syncthreads();
#pragma unroll
  for (int rr = 0; rr < 4; ++rr)
    PaT[((size_t)b * LB_ + (j0 + ty + rr * 8)) * LA_ + i0 + tx] = tile[tx][ty + rr * 8];
}

// ---------------- workspace layout (bytes) ----------------------------------
static const size_t OFF_A16  = 0;                    // 16 MiB f16 a
static const size_t OFF_B16  = OFF_A16 + 16777216;   // 16 MiB f16 b
static const size_t OFF_BT16 = OFF_B16 + 16777216;   // 16 MiB f16 b^T
static const size_t OFF_S    = OFF_BT16 + 16777216;  // 64 MiB f32 scores
static const size_t OFF_PB   = OFF_S + 67108864;     // 32 MiB f16 row-softmax
static const size_t OFF_PAT  = OFF_PB + 33554432;    // 32 MiB f16 col-softmax^T
static const size_t OFF_RMAX = OFF_PAT + 33554432;
static const size_t OFF_RSUM = OFF_RMAX + 65536;
static const size_t OFF_CMAX = OFF_RSUM + 65536;
static const size_t OFF_CSUM = OFF_CMAX + 65536;

extern "C" void kernel_launch(void* const* d_in, const int* in_sizes, int n_in,
                              void* d_out, int out_size, void* d_ws,
                              size_t ws_size, hipStream_t stream) {
  const float* a      = (const float*)d_in[0];
  const float* bmat   = (const float*)d_in[1];
  const int*   mask_a = (const int*)d_in[2];
  const int*   mask_b = (const int*)d_in[3];
  const float* temp   = (const float*)d_in[4];
  float* out = (float*)d_out;

  char* ws = (char*)d_ws;
  _Float16* a16  = (_Float16*)(ws + OFF_A16);
  _Float16* b16  = (_Float16*)(ws + OFF_B16);
  _Float16* bT16 = (_Float16*)(ws + OFF_BT16);
  float*    S    = (float*)(ws + OFF_S);
  _Float16* Pb   = (_Float16*)(ws + OFF_PB);
  _Float16* PaT  = (_Float16*)(ws + OFF_PAT);
  float* rmax = (float*)(ws + OFF_RMAX);
  float* rsum = (float*)(ws + OFF_RSUM);
  float* cmax = (float*)(ws + OFF_CMAX);
  float* csum = (float*)(ws + OFF_CSUM);

  const size_t nAB = (size_t)B_ * LA_ * H_;  // == B*LB*H
  const int n4 = (int)(nAB / 4);

  f32_to_f16<<<(n4 + 255) / 256, 256, 0, stream>>>(a, a16, n4);
  f32_to_f16<<<(n4 + 255) / 256, 256, 0, stream>>>(bmat, b16, n4);
  transpose_f16<<<dim3(H_ / 32, LB_ / 32, B_), dim3(32, 8), 0, stream>>>(
      bmat, bT16, LB_, H_);

  // S = a16 * bT16  (M=LA, N=LB, K=H) with temperature + mask epilogue
  gemm_wmma<0><<<dim3(LB_ / 128, LA_ / 128, B_), 256, 0, stream>>>(
      a16, bT16, S, LA_, LB_, H_, mask_a, mask_b, temp);

  row_stats<<<dim3(LA_, B_), 256, 0, stream>>>(S, rmax, rsum);
  col_stats<<<dim3(LB_ / 256, B_), 256, 0, stream>>>(S, cmax, csum);
  probs<<<dim3(LB_ / 32, LA_ / 32, B_), dim3(32, 8), 0, stream>>>(
      S, rmax, rsum, cmax, csum, Pb, PaT);

  // feature_a = Pb * b16   (M=LA, N=H, K=LB)
  gemm_wmma<1><<<dim3(H_ / 128, LA_ / 128, B_), 256, 0, stream>>>(
      Pb, b16, out, LA_, H_, LB_, nullptr, nullptr, nullptr);
  // feature_b = PaT * a16  (M=LB, N=H, K=LA)
  gemm_wmma<1><<<dim3(H_ / 128, LB_ / 128, B_), 256, 0, stream>>>(
      PaT, a16, out + nAB, LB_, H_, LA_, nullptr, nullptr, nullptr);
}
